// CDLSSTransform_17145509446106
// MI455X (gfx1250) — matrix operand order repaired and checked by reading, not compiled, not run on hardware
//
#include <hip/hip_runtime.h>
#include <hip/hip_bf16.h>

// ---------- WMMA vector types (CDNA5 / gfx1250) ----------
typedef __attribute__((ext_vector_type(16))) __bf16 v16bf;
typedef __attribute__((ext_vector_type(8)))  float  v8f;

#define D_BINS 59
#define C_OUTC 80
#define C_INC  256
#define FHT 32
#define FWT 88
#define IHT 256
#define IWT 704
#define NB  6
#define NPIX (NB*FHT*FWT)            // 16896
#define NXG 360
#define NYG 360
#define OUT_CH_STRIDE (NXG*NYG)      // 129600
#define OUT_B_STRIDE  (C_OUTC*OUT_CH_STRIDE)
#define L2PIX (NB*64*176)            // 67584 (dt2 output pixels)
#define MTILES 16                    // all GEMMs padded to 256 output rows

__device__ __forceinline__ unsigned short f2bf(float f) {
  unsigned u = __float_as_uint(f);
  u += 0x7FFFu + ((u >> 16) & 1u);        // round-to-nearest-even
  return (unsigned short)(u >> 16);
}

// ---------- per-row (per-channel) sum & sumsq ----------
__global__ void k_rowstats(const float* __restrict__ X, long L, float* __restrict__ out) {
  int c = blockIdx.x;
  const float* row = X + (long)c * L;
  float s = 0.f, q = 0.f;
  for (long i = threadIdx.x; i < L; i += blockDim.x) { float v = row[i]; s += v; q += v * v; }
  __shared__ float ss[256], qs[256];
  int t = threadIdx.x;
  ss[t] = s; qs[t] = q; __syncthreads();
  for (int o = 128; o > 0; o >>= 1) {
    if (t < o) { ss[t] += ss[t + o]; qs[t] += qs[t + o]; }
    __syncthreads();
  }
  if (t == 0) { out[2 * c] = ss[0]; out[2 * c + 1] = qs[0]; }
}

// ---------- BN(train) -> per-channel scale/shift ----------
__global__ void k_bnparams(const float* __restrict__ stats, const float* __restrict__ g,
                           const float* __restrict__ bt, float invL, int C,
                           float* __restrict__ scale, float* __restrict__ shift) {
  int c = blockIdx.x * blockDim.x + threadIdx.x;
  if (c >= C) return;
  float m  = stats[2 * c] * invL;
  float v  = stats[2 * c + 1] * invL - m * m;
  float rs = rsqrtf(v + 1e-5f);
  float sc = g[c] * rs;
  scale[c] = sc; shift[c] = bt[c] - m * sc;
}

// ---------- dt1 (1x1, 1ch input) folded into affine a*d + c ----------
__global__ void k_dt1params(const float* __restrict__ dstats, const float* __restrict__ w1,
                            const float* __restrict__ g1, const float* __restrict__ bt1,
                            float invL, float* __restrict__ ac) {
  int c = threadIdx.x; if (c >= 8) return;
  float m = dstats[0] * invL, q = dstats[1] * invL;
  float var = q - m * m;
  float w = w1[c];
  float rs = rsqrtf(w * w * var + 1e-5f);
  float a = g1[c] * w * rs;
  ac[c] = a; ac[8 + c] = bt1[c] - a * m;
}

// ---------- dt2: 5x5 s4 conv, dt1+BN+ReLU fused on the fly ----------
__global__ void k_dt2(const float* __restrict__ dI, const float* __restrict__ W2,
                      const float* __restrict__ ac, float* __restrict__ A2) {
  int l  = blockIdx.x * 256 + threadIdx.x;      // < 67584
  int co = blockIdx.y;
  int b  = l / (64 * 176); int r = l % (64 * 176);
  int oy = r / 176, ox = r % 176;
  float a[8], c[8];
#pragma unroll
  for (int i = 0; i < 8; i++) { a[i] = ac[i]; c[i] = ac[8 + i]; }
  const float* w = W2 + co * 8 * 25;            // OIHW [32][8][5][5]
  float sum = 0.f;
  for (int ky = 0; ky < 5; ++ky) {
    int iy = oy * 4 + ky - 2;
    if (iy < 0 || iy >= IHT) continue;
    for (int kx = 0; kx < 5; ++kx) {
      int ix = ox * 4 + kx - 2;
      if (ix < 0 || ix >= IWT) continue;
      float dv = dI[(b * IHT + iy) * IWT + ix];
#pragma unroll
      for (int ci = 0; ci < 8; ++ci) {
        float v = fmaxf(a[ci] * dv + c[ci], 0.f);
        sum += v * w[(ci * 5 + ky) * 5 + kx];
      }
    }
  }
  A2[co * L2PIX + l] = sum;
}

// ---------- dt3: 5x5 s2 conv, 32->64, BN+ReLU of A2 on the fly ----------
__global__ void k_dt3(const float* __restrict__ A2, const float* __restrict__ W3,
                      const float* __restrict__ sc, const float* __restrict__ sh,
                      float* __restrict__ A3) {
  int n  = blockIdx.x * 256 + threadIdx.x;      // < 16896
  int co = blockIdx.y;
  int b  = n / (FHT * FWT); int r = n % (FHT * FWT);
  int oy = r / FWT, ox = r % FWT;
  const float* w = W3 + co * 32 * 25;           // OIHW [64][32][5][5]
  float sum = 0.f;
  for (int ky = 0; ky < 5; ++ky) {
    int iy = oy * 2 + ky - 2;
    if (iy < 0 || iy >= 64) continue;
    for (int kx = 0; kx < 5; ++kx) {
      int ix = ox * 2 + kx - 2;
      if (ix < 0 || ix >= 176) continue;
      int base = (b * 64 + iy) * 176 + ix;
#pragma unroll 8
      for (int ci = 0; ci < 32; ++ci) {
        float v = fmaxf(sc[ci] * A2[ci * L2PIX + base] + sh[ci], 0.f);
        sum += v * w[(ci * 5 + ky) * 5 + kx];
      }
    }
  }
  A3[co * NPIX + n] = sum;
}

// ---------- pack Z0: concat(relu(bn(A3)), x) -> padded channels-last bf16 [B][34][90][320] ----------
__global__ void k_pack0(const float* __restrict__ A3, const float* __restrict__ x,
                        const float* __restrict__ sc, const float* __restrict__ sh,
                        unsigned short* __restrict__ Z0) {
  int site = blockIdx.x;                         // B*34*90
  int b = site / (34 * 90); int r = site % (34 * 90);
  int hp = r / 90, wp = r % 90;
  int h = hp - 1, w = wp - 1;
  bool interior = (h >= 0 && h < FHT && w >= 0 && w < FWT);
  unsigned short* dst = Z0 + site * 320;
  int n = (b * FHT + h) * FWT + w;
  for (int c = threadIdx.x; c < 320; c += blockDim.x) {
    float v = 0.f;
    if (interior) {
      if (c < 64) v = fmaxf(sc[c] * A3[c * NPIX + n] + sh[c], 0.f);
      else        v = x[((b * C_INC + (c - 64)) * FHT + h) * FWT + w];
    }
    dst[c] = f2bf(v);
  }
}

// ---------- pack Z1: relu(bn(C)) -> padded channels-last bf16 [B][34][90][256] ----------
__global__ void k_packmid(const float* __restrict__ C, const float* __restrict__ sc,
                          const float* __restrict__ sh, unsigned short* __restrict__ Z) {
  int site = blockIdx.x;
  int b = site / (34 * 90); int r = site % (34 * 90);
  int hp = r / 90, wp = r % 90;
  int h = hp - 1, w = wp - 1;
  bool interior = (h >= 0 && h < FHT && w >= 0 && w < FWT);
  int n = (b * FHT + h) * FWT + w;
  unsigned short* dst = Z + site * 256;
  for (int c = threadIdx.x; c < 256; c += blockDim.x) {
    float v = 0.f;
    if (interior) v = fmaxf(sc[c] * C[c * NPIX + n] + sh[c], 0.f);
    dst[c] = f2bf(v);
  }
}

// ---------- pack Z2: relu(bn(C2)) -> channels-last bf16 [NPIX][256] (no pad; 1x1 conv) ----------
__global__ void k_pack2(const float* __restrict__ C, const float* __restrict__ sc,
                        const float* __restrict__ sh, unsigned short* __restrict__ Z) {
  int n = blockIdx.x;
  for (int c = threadIdx.x; c < 256; c += blockDim.x)
    Z[n * 256 + c] = f2bf(fmaxf(sc[c] * C[c * NPIX + n] + sh[c], 0.f));
}

// ---------- repack OIHW weights into per-lane WMMA A-fragment layout ----------
// Apack[kt][mtile(16)][lane(32)][e(16)], K(lane,e) = kt*32 + (lane/16)*16 + e
__global__ void k_repackA(const float* __restrict__ W, unsigned short* __restrict__ Apack,
                          int Ktiles, int Cin, int KH, int KW, int Mreal) {
  int idx = blockIdx.x * blockDim.x + threadIdx.x;
  int total = Ktiles * MTILES * 32 * 16;
  if (idx >= total) return;
  int e    = idx & 15;
  int lane = (idx >> 4) & 31;
  int t    = idx >> 9;
  int mt = t & (MTILES - 1);
  int kt = t >> 4;
  int m  = mt * 16 + (lane & 15);
  int K  = kt * 32 + ((lane >> 4) << 4) + e;
  float v = 0.f;
  if (m < Mreal) {
    int kk = K / Cin, ci = K % Cin;
    int kh = kk / KW, kw = kk % KW;
    v = W[((m * Cin + ci) * KH + kh) * KW + kw];
  }
  Apack[idx] = f2bf(v);
}

// ---------- implicit-GEMM conv via v_wmma_f32_16x16x32_bf16 ----------
// C[M=256][NPIX] = W * im2col(Z), Z channels-last padded [B][Hp][Wp][Cin].
// 4 waves/block, wave tile 64x32, software-pipelined K loop (Ktiles even).
#define LOAD_FRAGS(AB, BB)                                                    \
  do {                                                                        \
    int abase_ = ktn * (MTILES * 32) + abase;                                 \
    _Pragma("unroll") for (int mt_ = 0; mt_ < MT; ++mt_)                      \
        AB[mt_] = Ap[abase_ + mt_ * 32];                                      \
    _Pragma("unroll") for (int nt_ = 0; nt_ < NT; ++nt_)                      \
        BB[nt_] = *(const v16bf*)(Bact + rowbase[nt_] + boff_n);              \
    ++ktn; ++cb_n; boff_n += 32;                                              \
    if (cb_n == CinT) { cb_n = 0; ++kk_n;                                     \
      boff_n = ((kk_n / KW) * Wp + (kk_n % KW)) * Cin + khalf; }              \
  } while (0)

#define MMA_STEP(AB, BB)                                                      \
  do {                                                                        \
    _Pragma("unroll") for (int mt_ = 0; mt_ < MT; ++mt_)                      \
      _Pragma("unroll") for (int nt_ = 0; nt_ < NT; ++nt_)                    \
        acc[mt_][nt_] = __builtin_amdgcn_wmma_f32_16x16x32_bf16(              \
            false, AB[mt_], false, BB[nt_], (short)0, acc[mt_][nt_],          \
            false, false);                                                    \
  } while (0)

template <int KH, int KW>
__global__ __launch_bounds__(128) void k_gemm_wmma(
    const unsigned short* __restrict__ Apack, const unsigned short* __restrict__ Bact,
    float* __restrict__ Cout, const float* __restrict__ bias,
    int Mreal, int Cin, int Hp, int Wp) {
  const int MT = 4, NT = 2;
  int lane = threadIdx.x & 31;
  int wave = threadIdx.x >> 5;
  int n0   = blockIdx.x * (NT * 16);
  int CinT = Cin >> 5;
  int Ktiles = KH * KW * CinT;

  int rowbase[NT];
#pragma unroll
  for (int nt = 0; nt < NT; ++nt) {
    int n = n0 + nt * 16 + (lane & 15);
    int b = n / (FHT * FWT); int r = n % (FHT * FWT);
    int h = r / FWT, w = r % FWT;
    rowbase[nt] = ((b * Hp + h) * Wp + w) * Cin;
  }
  int khalf = (lane >> 4) << 4;                  // 0 or 16
  int abase = wave * (MT * 32) + lane;           // v16bf units within one kt slab

  v8f zero = {0.f, 0.f, 0.f, 0.f, 0.f, 0.f, 0.f, 0.f};
  v8f acc[MT][NT];
#pragma unroll
  for (int i = 0; i < MT; i++)
#pragma unroll
    for (int j = 0; j < NT; j++) acc[i][j] = zero;

  const v16bf* Ap = (const v16bf*)Apack;
  v16bf a0[MT], a1[MT], b0[NT], b1[NT];
  int ktn = 0, kk_n = 0, cb_n = 0, boff_n = khalf;

  LOAD_FRAGS(a0, b0);                            // kt = 0
  for (int kt = 0; kt < Ktiles; kt += 2) {
    LOAD_FRAGS(a1, b1);                          // kt+1 (Ktiles even -> valid)
    __builtin_prefetch((const void*)(Ap + ktn * (MTILES * 32) + abase), 0, 1);
    MMA_STEP(a0, b0);
    if (kt + 2 < Ktiles) LOAD_FRAGS(a0, b0);     // kt+2 (uniform branch)
    MMA_STEP(a1, b1);
  }

  // store: M = mtile*16 + (lane/16)*8 + r ; N = n0 + nt*16 + lane%16
#pragma unroll
  for (int mt = 0; mt < MT; ++mt) {
#pragma unroll
    for (int nt = 0; nt < NT; ++nt) {
      int ncol = n0 + nt * 16 + (lane & 15);
#pragma unroll
      for (int r = 0; r < 8; ++r) {
        int m = (wave * MT + mt) * 16 + ((lane >> 4) << 3) + r;
        if (m < Mreal) {
          float v = acc[mt][nt][r];
          if (bias) v += bias[m];
          Cout[m * NPIX + ncol] = v;
        }
      }
    }
  }
}

// ---------- softmax over 59 depth logits per pixel ----------
__global__ void k_softmax(const float* __restrict__ C3, float* __restrict__ P) {
  int n = blockIdx.x * blockDim.x + threadIdx.x;   // 66*256 = 16896
  float mx = -1e30f;
  for (int m = 0; m < D_BINS; ++m) mx = fmaxf(mx, C3[m * NPIX + n]);
  float s = 0.f;
  for (int m = 0; m < D_BINS; ++m) s += __expf(C3[m * NPIX + n] - mx);
  float inv = 1.f / s;
  for (int m = 0; m < D_BINS; ++m) P[m * NPIX + n] = __expf(C3[m * NPIX + n] - mx) * inv;
}

// ---------- geometry: per-batch 3x3 inverses & combine matrices ----------
__device__ void inv3(const float* A, float* o) {
  float a=A[0],b=A[1],c=A[2],d=A[3],e=A[4],f=A[5],g=A[6],h=A[7],i=A[8];
  float A00=e*i-f*h, A01=c*h-b*i, A02=b*f-c*e;
  float A10=f*g-d*i, A11=a*i-c*g, A12=c*d-a*f;
  float A20=d*h-e*g, A21=b*g-a*h, A22=a*e-b*d;
  float id = 1.f / (a*A00 + b*A10 + c*A20);
  o[0]=A00*id; o[1]=A01*id; o[2]=A02*id;
  o[3]=A10*id; o[4]=A11*id; o[5]=A12*id;
  o[6]=A20*id; o[7]=A21*id; o[8]=A22*id;
}

__global__ void k_geomsetup(const float* __restrict__ c2l_rots, const float* __restrict__ c2l_trans,
                            const float* __restrict__ intrins, const float* __restrict__ post_rots,
                            const float* __restrict__ post_trans, float* __restrict__ G) {
  int b = threadIdx.x; if (b >= NB) return;
  float invP[9], invK[9];
  inv3(post_rots + b * 9, invP);
  inv3(intrins + b * 9, invK);
  const float* R = c2l_rots + b * 9;
  float* g = G + b * 24;
  for (int i = 0; i < 9; i++) g[i] = invP[i];
  for (int i = 0; i < 3; i++)
    for (int j = 0; j < 3; j++) {
      float s = 0.f;
      for (int k = 0; k < 3; k++) s += R[i * 3 + k] * invK[k * 3 + j];
      g[9 + i * 3 + j] = s;
    }
  for (int i = 0; i < 3; i++) { g[18 + i] = post_trans[b * 3 + i]; g[21 + i] = c2l_trans[b * 3 + i]; }
}

// ---------- geometry: voxel index per (d, pixel); -1 if outside grid ----------
__global__ void k_geometry(const float* __restrict__ G, int* __restrict__ V) {
  int idx = blockIdx.x * blockDim.x + threadIdx.x;
  if (idx >= D_BINS * NPIX) return;
  int dbin = idx / NPIX; int n = idx % NPIX;
  int b = n / (FHT * FWT); int r = n % (FHT * FWT);
  int h = r / FWT, w = r % FWT;
  const float* g = G + b * 24;
  float xs = w * (703.f / 87.f);
  float ys = h * (255.f / 31.f);
  float zs = dbin + 1.f;
  float p0 = xs - g[18], p1 = ys - g[19], p2 = zs - g[20];
  float q0 = g[0]*p0 + g[1]*p1 + g[2]*p2;
  float q1 = g[3]*p0 + g[4]*p1 + g[5]*p2;
  float q2 = g[6]*p0 + g[7]*p1 + g[8]*p2;
  q0 *= q2; q1 *= q2;
  float r0 = g[9]*q0  + g[10]*q1 + g[11]*q2 + g[21];
  float r1 = g[12]*q0 + g[13]*q1 + g[14]*q2 + g[22];
  float r2 = g[15]*q0 + g[16]*q1 + g[17]*q2 + g[23];
  int gx = (int)((r0 + 54.f) / 0.3f);   // trunc toward zero, matches .astype(int32)
  int gy = (int)((r1 + 54.f) / 0.3f);
  int gz = (int)((r2 + 10.f) / 20.f);
  bool kept = (gx >= 0) & (gx < NXG) & (gy >= 0) & (gy < NYG) & (gz == 0);
  V[idx] = kept ? (b * OUT_B_STRIDE + gx * NYG + gy) : -1;
}

// ---------- splat: out[b,c,gx,gy] += softmax_d * feat_c (segment-sum via atomics) ----------
__global__ void k_splat(const float* __restrict__ C3, const float* __restrict__ P,
                        const int* __restrict__ V, float* __restrict__ out) {
  int n = blockIdx.x;
  int c = threadIdx.x;                            // blockDim = 80
  float feat = C3[(D_BINS + c) * NPIX + n];
  for (int dd = 0; dd < D_BINS; ++dd) {
    int v = V[dd * NPIX + n];
    if (v >= 0) {
      float p = P[dd * NPIX + n];
      atomicAdd(out + v + c * OUT_CH_STRIDE, p * feat);
    }
  }
}

// ============================ host ============================
static inline size_t alignup(size_t x) { return (x + 255) & ~(size_t)255; }

extern "C" void kernel_launch(void* const* d_in, const int* in_sizes, int n_in,
                              void* d_out, int out_size, void* d_ws, size_t ws_size,
                              hipStream_t stream) {
  const float* xin   = (const float*)d_in[0];
  const float* dimg  = (const float*)d_in[1];
  const float* c2l_r = (const float*)d_in[2];
  const float* c2l_t = (const float*)d_in[3];
  const float* intr  = (const float*)d_in[4];
  const float* prot  = (const float*)d_in[5];
  const float* ptrn  = (const float*)d_in[6];

  // Resolve params dict flattening: insertion order (in_sizes[7]==8 -> dt1_w) vs alphabetical pytree.
  const float* P_[22];
  if (n_in >= 29 && in_sizes[7] == 8) {
    for (int i = 0; i < 22; i++) P_[i] = (const float*)d_in[7 + i];
  } else {
    static const int mp[22] = {13,15,14,12, 17,19,18,16, 21,20, 1,3,2,0, 5,7,6,4, 9,11,10,8};
    for (int i = 0; i < 22; i++) P_[mp[i]] = (const float*)d_in[7 + i];
  }
  const float *dt1_w=P_[0],              *dt1_g=P_[2],  *dt1_bt=P_[3];
  const float *dt2_w=P_[4],              *dt2_g=P_[6],  *dt2_bt=P_[7];
  const float *dt3_w=P_[8],              *dt3_g=P_[10], *dt3_bt=P_[11];
  const float *dn1_w=P_[12],             *dn1_g=P_[14], *dn1_bt=P_[15];
  const float *dn2_w=P_[16],             *dn2_g=P_[18], *dn2_bt=P_[19];
  const float *dn3_w=P_[20], *dn3_b=P_[21];

  // workspace bump allocator
  char* ws = (char*)d_ws;
  size_t off = 0;
  auto alloc = [&](size_t bytes) -> void* { void* p = ws + off; off = alignup(off + bytes); return p; };
  float* G      = (float*)alloc(NB * 24 * 4);
  float* dstats = (float*)alloc(2 * 4);
  float* dt1ac  = (float*)alloc(16 * 4);
  float* A2     = (float*)alloc((size_t)32 * L2PIX * 4);
  float* st2    = (float*)alloc(64 * 4);
  float* sc2    = (float*)alloc(32 * 4);
  float* sh2    = (float*)alloc(32 * 4);
  float* A3     = (float*)alloc((size_t)64 * NPIX * 4);
  float* st3    = (float*)alloc(128 * 4);
  float* sc3    = (float*)alloc(64 * 4);
  float* sh3    = (float*)alloc(64 * 4);
  unsigned short* Z0  = (unsigned short*)alloc((size_t)NB * 34 * 90 * 320 * 2);
  unsigned short* Ap1 = (unsigned short*)alloc((size_t)90 * MTILES * 512 * 2);
  float* C1     = (float*)alloc((size_t)256 * NPIX * 4);
  float* st4    = (float*)alloc(512 * 4);
  float* sc4    = (float*)alloc(256 * 4);
  float* sh4    = (float*)alloc(256 * 4);
  unsigned short* Z1  = (unsigned short*)alloc((size_t)NB * 34 * 90 * 256 * 2);
  unsigned short* Ap2 = (unsigned short*)alloc((size_t)72 * MTILES * 512 * 2);
  float* C2     = (float*)alloc((size_t)256 * NPIX * 4);
  float* st5    = (float*)alloc(512 * 4);
  float* sc5    = (float*)alloc(256 * 4);
  float* sh5    = (float*)alloc(256 * 4);
  unsigned short* Z2  = (unsigned short*)alloc((size_t)NPIX * 256 * 2);
  unsigned short* Ap3 = (unsigned short*)alloc((size_t)8 * MTILES * 512 * 2);
  float* C3     = (float*)alloc((size_t)144 * NPIX * 4);
  float* Psm    = (float*)alloc((size_t)D_BINS * NPIX * 4);
  int*   Vox    = (int*)alloc((size_t)D_BINS * NPIX * 4);
  (void)ws_size; (void)out_size;

  hipMemsetAsync(d_out, 0, (size_t)out_size * sizeof(float), stream);

  // geometry (independent of features)
  k_geomsetup<<<1, 32, 0, stream>>>(c2l_r, c2l_t, intr, prot, ptrn, G);
  k_geometry<<<(D_BINS * NPIX + 255) / 256, 256, 0, stream>>>(G, Vox);

  // dt1 stats from d directly (1x1 conv of 1 channel)
  const long Ld = (long)NB * IHT * IWT;
  k_rowstats<<<1, 256, 0, stream>>>(dimg, Ld, dstats);
  k_dt1params<<<1, 32, 0, stream>>>(dstats, dt1_w, dt1_g, dt1_bt, 1.f / (float)Ld, dt1ac);

  // dt2, dt3 (VALU convs, bandwidth-trivial)
  k_dt2<<<dim3(L2PIX / 256, 32), 256, 0, stream>>>(dimg, dt2_w, dt1ac, A2);
  k_rowstats<<<32, 256, 0, stream>>>(A2, L2PIX, st2);
  k_bnparams<<<1, 64, 0, stream>>>(st2, dt2_g, dt2_bt, 1.f / (float)L2PIX, 32, sc2, sh2);
  k_dt3<<<dim3(NPIX / 256, 64), 256, 0, stream>>>(A2, dt3_w, sc2, sh2, A3);
  k_rowstats<<<64, 256, 0, stream>>>(A3, NPIX, st3);
  k_bnparams<<<1, 64, 0, stream>>>(st3, dt3_g, dt3_bt, 1.f / (float)NPIX, 64, sc3, sh3);

  // dn1: WMMA implicit GEMM (M=256, K=2880, N=16896)
  k_pack0<<<NB * 34 * 90, 128, 0, stream>>>(A3, xin, sc3, sh3, Z0);
  k_repackA<<<(90 * MTILES * 512) / 256, 256, 0, stream>>>(dn1_w, Ap1, 90, 320, 3, 3, 256);
  k_gemm_wmma<3, 3><<<NPIX / 32, 128, 0, stream>>>(Ap1, Z0, C1, nullptr, 256, 320, 34, 90);
  k_rowstats<<<256, 256, 0, stream>>>(C1, NPIX, st4);
  k_bnparams<<<1, 256, 0, stream>>>(st4, dn1_g, dn1_bt, 1.f / (float)NPIX, 256, sc4, sh4);

  // dn2: WMMA implicit GEMM (M=256, K=2304, N=16896)
  k_packmid<<<NB * 34 * 90, 256, 0, stream>>>(C1, sc4, sh4, Z1);
  k_repackA<<<(72 * MTILES * 512) / 256, 256, 0, stream>>>(dn2_w, Ap2, 72, 256, 3, 3, 256);
  k_gemm_wmma<3, 3><<<NPIX / 32, 128, 0, stream>>>(Ap2, Z1, C2, nullptr, 256, 256, 34, 90);
  k_rowstats<<<256, 256, 0, stream>>>(C2, NPIX, st5);
  k_bnparams<<<1, 256, 0, stream>>>(st5, dn2_g, dn2_bt, 1.f / (float)NPIX, 256, sc5, sh5);

  // dn3: 1x1 WMMA GEMM (M=139 padded to 256, K=256) with bias
  k_pack2<<<NPIX, 256, 0, stream>>>(C2, sc5, sh5, Z2);
  k_repackA<<<(8 * MTILES * 512) / 256, 256, 0, stream>>>(dn3_w, Ap3, 8, 256, 1, 1, 139);
  k_gemm_wmma<1, 1><<<NPIX / 32, 128, 0, stream>>>(Ap3, Z2, C3, dn3_b, 139, 256, FHT, FWT);

  // softmax over depth, then scatter-sum into BEV grid
  k_softmax<<<NPIX / 256, 256, 0, stream>>>(C3, Psm);
  k_splat<<<NPIX, 80, 0, stream>>>(C3, Psm, Vox, (float*)d_out);
}
